// RnnGenerator_4569845203004
// MI455X (gfx1250) — matrix operand order repaired and checked by reading, not compiled, not run on hardware
//
#include <hip/hip_runtime.h>
#include <hip/hip_bf16.h>

typedef __attribute__((ext_vector_type(16))) __bf16 v16bf;
typedef __attribute__((ext_vector_type(8)))  __bf16 v8bf;
typedef __attribute__((ext_vector_type(8)))  float  v8f;

#define CDIM   2048
#define LAYERS 4
#define STEPS  512
#define TILES  128   // CDIM/16
#define CHUNKS 64    // CDIM/32
#define GBLK   32    // persistent-kernel blocks (must all be co-resident)
#define BTHR   128   // 4 waves/block -> 128 waves total = 128 tiles

// ---------------------------------------------------------------- grid barrier
__device__ __forceinline__ void grid_sync(unsigned* cnt, unsigned nblocks) {
    __syncthreads();
    if (threadIdx.x == 0) {
        __threadfence();
        unsigned ticket = atomicAdd(cnt, 1u);
        unsigned target = (ticket / nblocks + 1u) * nblocks;
        while (__hip_atomic_load(cnt, __ATOMIC_RELAXED, __HIP_MEMORY_SCOPE_AGENT) < target) {
            __builtin_amdgcn_s_sleep(1);
        }
        __threadfence();
    }
    __syncthreads();
}

// ---------------------------------------------------------------- init state
__global__ void init_state(const float* __restrict__ x, __bf16* __restrict__ hbf,
                           float* __restrict__ ss, unsigned* __restrict__ cnt) {
    int g = blockIdx.x * blockDim.x + threadIdx.x;
    if (g == 0) *cnt = 0u;
    if (g < 8) ss[g] = 0.0f;
    for (int i = g; i < LAYERS * CDIM; i += gridDim.x * blockDim.x) {
        // hid0: layer 0 = x, layers 1..3 = 0 ; inp0 == hbf[3] == 0
        hbf[i] = (i < CDIM) ? (__bf16)x[i] : (__bf16)0.0f;
    }
}

// ------------------------------------------- weight swizzle f32 -> bf16 B-frags
// out[((layer*128 + tile)*64 + chunk)*32 + lane][e] = W[layer][16*tile + (lane&15)]
//                                                      [32*chunk + 8*(lane>>4) + (e&7) + 16*(e>>3)]
__global__ void swizzle_w(const float* __restrict__ W, __bf16* __restrict__ out) {
    const size_t total = (size_t)LAYERS * CDIM * CDIM;
    for (size_t o = blockIdx.x * (size_t)blockDim.x + threadIdx.x; o < total;
         o += (size_t)gridDim.x * blockDim.x) {
        const int e     = (int)(o & 15);
        const int lane  = (int)((o >> 4) & 31);
        const int c     = (int)((o >> 9) & 63);
        const int t     = (int)((o >> 15) & 127);
        const int layer = (int)(o >> 22);
        const int j  = t * 16 + (lane & 15);
        const int kk = c * 32 + ((lane >> 4) << 3) + (e & 7) + ((e >> 3) << 4);
        out[o] = (__bf16)W[((size_t)layer * CDIM + j) * CDIM + kk];
    }
}

// ---------------------------------------------------------------- LengthProducer
__global__ void lp_matvec(const float* __restrict__ W, const float* __restrict__ b,
                          const float* __restrict__ x, float* __restrict__ y) {
    const int lane = threadIdx.x & 31;
    const int j = blockIdx.x * (blockDim.x >> 5) + (threadIdx.x >> 5);
    const float* row = W + (size_t)j * CDIM;
    float s = 0.0f;
    for (int k = lane; k < CDIM; k += 32) s += row[k] * x[k];
#pragma unroll
    for (int m = 16; m; m >>= 1) s += __shfl_xor(s, m, 32);
    if (lane == 0) {
        float v = s + b[j];
        y[j] = v > 0.0f ? v : 0.2f * v;   // LeakyReLU(0.2)
    }
}

__global__ void lp_final(const float* __restrict__ wout, const float* __restrict__ bout,
                         const float* __restrict__ y, float* __restrict__ out) {
    __shared__ float red[8];
    float s = 0.0f;
    for (int k = threadIdx.x; k < CDIM; k += 256) s += wout[k] * y[k];
#pragma unroll
    for (int m = 16; m; m >>= 1) s += __shfl_xor(s, m, 32);
    if ((threadIdx.x & 31) == 0) red[threadIdx.x >> 5] = s;
    __syncthreads();
    if (threadIdx.x == 0) {
        float t = 0.0f;
#pragma unroll
        for (int i = 0; i < 8; ++i) t += red[i];
        t = fabsf(t + bout[0]);
        out[0] = fminf(t, 0.9999f);       // clamp(|.|, 0, 0.9999)
    }
}

// ---------------------------------------------------------------- persistent RNN
__global__ void __launch_bounds__(BTHR, 1)
rnn_kernel(const float* __restrict__ b_ih, const float* __restrict__ b_hh,
           const __bf16* __restrict__ wih, const __bf16* __restrict__ whh,
           float* __restrict__ h,       // [4][C] raw f32 outputs this step
           __bf16* __restrict__ hbf,    // [4][C] normalized carry (hbf[3] == inp)
           __bf16* __restrict__ hraw,   // [4][C] raw bf16 outputs this step
           float*  __restrict__ ss,     // [2][4] sum-of-squares (parity buffered)
           unsigned* __restrict__ cnt,
           float* __restrict__ seq)     // [512][C]
{
    const int lane = threadIdx.x & 31;
    const int t    = blockIdx.x * (BTHR / 32) + (threadIdx.x >> 5);  // tile 0..127
    const int g    = blockIdx.x * BTHR + threadIdx.x;                // 0..4095
    const int aoff = (lane & 16) ? 8 : 0;   // ISA 16-bit A-matrix lane K-offset

    for (int step = 0; step < STEPS; ++step) {
        const int par = step & 1;
        for (int layer = 0; layer < LAYERS; ++layer) {
            // layer input: step input (normalized h3) for layer 0, else raw h[l-1]
            const __bf16* a_in = (layer == 0) ? (hbf + 3 * CDIM)
                                              : (hraw + (layer - 1) * CDIM);
            const __bf16* a_h  = hbf + layer * CDIM;   // normalized prev hidden
            const __bf16* bi = wih + ((size_t)layer * TILES + t) * (CHUNKS * 512) + lane * 16;
            const __bf16* bh = whh + ((size_t)layer * TILES + t) * (CHUNKS * 512) + lane * 16;

            v8f acc = {};
#pragma unroll 2
            for (int c = 0; c < CHUNKS; ++c) {
                const int ab = c * 32 + aoff;
                v8bf ai_lo = *(const v8bf*)(a_in + ab);
                v8bf ai_hi = *(const v8bf*)(a_in + ab + 16);
                v8bf ah_lo = *(const v8bf*)(a_h + ab);
                v8bf ah_hi = *(const v8bf*)(a_h + ab + 16);
                v16bf bfi = *(const v16bf*)(bi + (size_t)c * 512);
                v16bf bfh = *(const v16bf*)(bh + (size_t)c * 512);
                v16bf ai, ah;
#pragma unroll
                for (int i = 0; i < 8; ++i) {
                    ai[i] = ai_lo[i]; ai[8 + i] = ai_hi[i];
                    ah[i] = ah_lo[i]; ah[8 + i] = ah_hi[i];
                }
                acc = __builtin_amdgcn_wmma_f32_16x16x32_bf16(
                          false, ai, false, bfi, (short)0, acc, false, false);
                acc = __builtin_amdgcn_wmma_f32_16x16x32_bf16(
                          false, ah, false, bfh, (short)0, acc, false, false);
            }
            // D row M=0 lives in acc[0] of lanes 0..15 (N = lane)
            float y = 0.0f;
            if (lane < 16) {
                const int j = t * 16 + lane;
                float v = acc[0] + b_ih[layer * CDIM + j] + b_hh[layer * CDIM + j];
                v = fmaxf(v, 0.0f);                 // ReLU
                h[layer * CDIM + j]    = v;
                hraw[layer * CDIM + j] = (__bf16)v;
                y = v;
            }
            float p = y * y;                        // lanes >= 16 contribute 0
#pragma unroll
            for (int m = 16; m; m >>= 1) p += __shfl_xor(p, m, 32);
            if (lane == 0) atomicAdd(&ss[par * 4 + layer], p);
            grid_sync(cnt, GBLK);
        }
        // ------- normalize carries, emit seq[step] (8192 elems over 4096 threads)
#pragma unroll
        for (int half = 0; half < 2; ++half) {
            const int e   = g + half * 4096;
            const int vec = e >> 11;
            const int i   = e & (CDIM - 1);
            const float val = h[vec * CDIM + i];
            const float sc  = 1.0f / (sqrtf(ss[par * 4 + vec]) + 1e-12f);
            const float nv  = val * sc;
            hbf[vec * CDIM + i] = (__bf16)nv;
            if (vec == 3) seq[(size_t)step * CDIM + i] = nv;   // inp == norm(h3)
        }
        if (g == 0) {   // reset the other parity's ss for step+1 (safe: barriered)
            ss[(1 - par) * 4 + 0] = 0.0f; ss[(1 - par) * 4 + 1] = 0.0f;
            ss[(1 - par) * 4 + 2] = 0.0f; ss[(1 - par) * 4 + 3] = 0.0f;
        }
        grid_sync(cnt, GBLK);
    }
}

// ---------------------------------------------------------------- launcher
extern "C" void kernel_launch(void* const* d_in, const int* in_sizes, int n_in,
                              void* d_out, int out_size, void* d_ws, size_t ws_size,
                              hipStream_t stream) {
    (void)in_sizes; (void)n_in; (void)out_size; (void)ws_size;
    const float* x       = (const float*)d_in[0];
    const float* lp_w    = (const float*)d_in[1];
    const float* lp_b    = (const float*)d_in[2];
    const float* lp_wout = (const float*)d_in[3];
    const float* lp_bout = (const float*)d_in[4];
    const float* w_ih    = (const float*)d_in[5];
    const float* b_ih    = (const float*)d_in[6];
    const float* w_hh    = (const float*)d_in[7];
    const float* b_hh    = (const float*)d_in[8];
    float* out = (float*)d_out;

    char* ws = (char*)d_ws;
    unsigned* cnt  = (unsigned*)(ws + 0);
    float*    ss   = (float*)(ws + 64);                    // [2][4]
    float*    h    = (float*)(ws + 256);                   // 32 KB
    __bf16*   hbf  = (__bf16*)(ws + 256 + 4 * CDIM * 4);   // 16 KB
    __bf16*   hraw = (__bf16*)(ws + 256 + 4 * CDIM * 4 + 4 * CDIM * 2);  // 16 KB
    float*    lpy0 = (float*)(ws + 65792);                 // 8 KB
    float*    lpy1 = (float*)(ws + 73984);                 // 8 KB
    __bf16* wih_swz = (__bf16*)(ws + 131072);              // 33.5 MB
    __bf16* whh_swz = wih_swz + (size_t)LAYERS * CDIM * CDIM;  // +33.5 MB

    init_state<<<8, 256, 0, stream>>>(x, hbf, ss, cnt);
    swizzle_w<<<4096, 256, 0, stream>>>(w_ih, wih_swz);
    swizzle_w<<<4096, 256, 0, stream>>>(w_hh, whh_swz);

    // LengthProducer: 3 leaky-relu matvecs + |dot| clamp -> out[512*2048]
    lp_matvec<<<256, 256, 0, stream>>>(lp_w + 0 * (size_t)CDIM * CDIM, lp_b + 0 * CDIM, x,    lpy0);
    lp_matvec<<<256, 256, 0, stream>>>(lp_w + 1 * (size_t)CDIM * CDIM, lp_b + 1 * CDIM, lpy0, lpy1);
    lp_matvec<<<256, 256, 0, stream>>>(lp_w + 2 * (size_t)CDIM * CDIM, lp_b + 2 * CDIM, lpy1, lpy0);
    lp_final<<<1, 256, 0, stream>>>(lp_wout, lp_bout, lpy0, out + (size_t)STEPS * CDIM);

    rnn_kernel<<<GBLK, BTHR, 0, stream>>>(b_ih, b_hh, wih_swz, whh_swz,
                                          h, hbf, hraw, ss, cnt, out);
}
